// BRASKModel_72129680769774
// MI455X (gfx1250) — compile-verified, matrix operand order checked by reading, not run on hardware
//
#include <hip/hip_runtime.h>
#include <hip/hip_bf16.h>

typedef __bf16 bf16;
typedef __attribute__((ext_vector_type(8)))  bf16  v8bf;
typedef __attribute__((ext_vector_type(16))) bf16  v16bf;
typedef __attribute__((ext_vector_type(8)))  float v8f;

// ---------------- wave helpers (wave32) ----------------
__device__ __forceinline__ float wsum(float s){
#pragma unroll
  for (int off = 16; off > 0; off >>= 1) s += __shfl_xor(s, off, 32);
  return s;
}
__device__ __forceinline__ float wmax(float s){
#pragma unroll
  for (int off = 16; off > 0; off >>= 1) s = fmaxf(s, __shfl_xor(s, off, 32));
  return s;
}

// ---------------- WMMA fragment load (unguarded) ----------------
// 16-bit A layout (ISA 7.12.2): lanes 0-15 row M=lane, elems {0..7}=K0..7,
// {8..15}=K16..23 ; lanes 16-31 row M=lane-16, K offset +8. B mirrored from
// Bt stored [N x K] row-major. `p` is the per-lane base (row, k-half applied).
__device__ __forceinline__ v16bf ldfrag(const bf16* p){
  const v8bf* q = (const v8bf*)p;
  v8bf lo = q[0];        // K offsets {0..7}
  v8bf hi = q[2];        // K offsets {16..23}
  return __builtin_shufflevector(lo, hi, 0,1,2,3,4,5,6,7,8,9,10,11,12,13,14,15);
}

#define WMMA_BF16(A_, B_, C_) \
  __builtin_amdgcn_wmma_f32_16x16x32_bf16(false, (A_), false, (B_), (short)0, (C_), false, false)

// ---------------- WMMA GEMM: wave computes a 16x64 tile ----------------
// Out[M,N] = A[M,K](bf16,rm) * Bt[N,K](bf16,rm)^T + bias[N]
// Requires: M % 16 == 0, N % 64 == 0, K % 32 == 0 (true for all launches).
__global__ void wmma_gemm4(const bf16* __restrict__ A, const bf16* __restrict__ Bt,
                           const float* __restrict__ bias, float* __restrict__ Out,
                           int M, int N, int K,
                           long strideA, long strideB, long strideO){
  int lane  = threadIdx.x & 31;
  int wave  = threadIdx.x >> 5;
  int ncg   = N >> 6;                 // 64-wide column groups
  int nmt   = M >> 4;
  int tile  = blockIdx.x * 8 + wave;
  if (tile >= nmt * ncg) return;
  int tm = tile / ncg, tn = tile % ncg;

  int lr   = lane & 15;               // lane row within fragment
  int khof = (lane >> 4) * 8;         // k-half offset

  const bf16* Ap = A  + (size_t)blockIdx.y * strideA + (size_t)(tm * 16 + lr) * K + khof;
  const bf16* Bp = Bt + (size_t)blockIdx.y * strideB + (size_t)(tn * 64 + lr) * K + khof;
  float*      Ob = Out + (size_t)blockIdx.y * strideO;

  v8f a0 = {0.f,0.f,0.f,0.f,0.f,0.f,0.f,0.f};
  v8f a1 = a0, a2 = a0, a3 = a0;
  const size_t bstep = (size_t)16 * K;     // 16 N-rows of Bt

  for (int k0 = 0; k0 < K; k0 += 32){
    v16bf fa  = ldfrag(Ap + k0);
    v16bf fb0 = ldfrag(Bp + k0);
    v16bf fb1 = ldfrag(Bp + bstep + k0);
    v16bf fb2 = ldfrag(Bp + 2 * bstep + k0);
    v16bf fb3 = ldfrag(Bp + 3 * bstep + k0);
    a0 = WMMA_BF16(fa, fb0, a0);
    a1 = WMMA_BF16(fa, fb1, a1);
    a2 = WMMA_BF16(fa, fb2, a2);
    a3 = WMMA_BF16(fa, fb3, a3);
  }

  int row0 = tm * 16 + ((lane >> 4) << 3);
  int col0 = tn * 64 + lr;
  v8f accs[4] = {a0, a1, a2, a3};
#pragma unroll
  for (int s = 0; s < 4; ++s){
    int col = col0 + s * 16;
    float bv = bias ? bias[col] : 0.f;
    float* o = Ob + (size_t)row0 * N + col;
#pragma unroll
    for (int r = 0; r < 8; ++r) o[(size_t)r * N] = accs[s][r] + bv;
  }
}

// ---------------- small kernels ----------------
__global__ void rowdot(const float* __restrict__ X, const float* __restrict__ w,
                       const float* __restrict__ bias, float* __restrict__ out,
                       int nrows, int H){
  int wv = (int)((blockIdx.x * blockDim.x + threadIdx.x) >> 5);
  int lane = threadIdx.x & 31;
  if (wv >= nrows) return;
  const float* xr = X + (size_t)wv * H;
  float s = 0.f;
  for (int h = lane; h < H; h += 32) s += xr[h] * w[h];
  s = wsum(s);
  if (lane == 0) out[wv] = s + (bias ? bias[0] : 0.f);
}

__global__ void rowdot2(const float* __restrict__ X, const float* __restrict__ Y,
                        const float* __restrict__ w, float* __restrict__ out,
                        int nrows, int H){
  int wv = (int)((blockIdx.x * blockDim.x + threadIdx.x) >> 5);
  int lane = threadIdx.x & 31;
  if (wv >= nrows) return;
  const float* xr = X + (size_t)wv * H;
  const float* yr = Y + (size_t)wv * H;
  float s = 0.f;
  for (int h = lane; h < H; h += 32) s += (xr[h] + yr[h]) * w[h];
  s = wsum(s);
  if (lane == 0) out[wv] = s;
}

__global__ void rel_proj(const float* __restrict__ T, const float* __restrict__ W,
                         const float* __restrict__ bias, float* __restrict__ out,
                         int R, int D, int H){
  int idx = blockIdx.x * blockDim.x + threadIdx.x;
  if (idx >= R * H) return;
  int h = idx % H, r = idx / H;
  float s = bias[h];
  for (int d = 0; d < D; ++d) s += T[r * D + d] * W[(size_t)d * H + h];
  out[idx] = s;
}

__global__ void to_bf16(const float* __restrict__ x, bf16* __restrict__ y, int n){
  int i = blockIdx.x * blockDim.x + threadIdx.x;
  if (i < n) y[i] = (bf16)x[i];
}

// convert n valid elements, zero-pad up to ntot (row padding for M%16 != 0)
__global__ void to_bf16_pad(const float* __restrict__ x, bf16* __restrict__ y,
                            int n, int ntot){
  int i = blockIdx.x * blockDim.x + threadIdx.x;
  if (i >= ntot) return;
  y[i] = (i < n) ? (bf16)x[i] : (bf16)0.0f;
}

__global__ void wt_bf16(const float* __restrict__ W, bf16* __restrict__ Wt, int H){
  size_t base = (size_t)blockIdx.y * H * H;
  int idx = blockIdx.x * blockDim.x + threadIdx.x;
  if (idx >= H * H) return;
  int k = idx % H, n = idx / H;
  Wt[base + (size_t)n * H + k] = (bf16)W[base + (size_t)k * H + n];
}

__global__ void tokT_bf16_k(const float* __restrict__ tok, bf16* __restrict__ out,
                            int B, int L, int H){
  int idx = blockIdx.x * blockDim.x + threadIdx.x;
  if (idx >= B * H * L) return;
  int l = idx % L;
  int h = (idx / L) % H;
  int b = idx / (L * H);
  out[idx] = (bf16)tok[((size_t)b * L + l) * H + h];
}

__global__ void span_extract(const float* __restrict__ fs, const float* __restrict__ fe,
                             const float* __restrict__ bs, const float* __restrict__ be,
                             int* __restrict__ meta, int B, int L){
  int t = threadIdx.x;
  if (t >= 2 * B) return;
  int head = t / B, b = t % B;
  const float* S = head ? bs : fs;
  const float* E = head ? be : fe;
  int sidx = 0, has = 0;
  for (int l = 0; l < L; ++l){ if (S[b * L + l] > 0.f){ sidx = l; has = 1; break; } }
  int eidx = sidx;
  for (int l = sidx; l < L; ++l){ if (E[b * L + l] > 0.f){ eidx = l; break; } }
  int* m = meta + t * 4;
  m[0] = sidx; m[1] = eidx - sidx + 1; m[2] = has;
}

__global__ void pad_build(const float* __restrict__ tok, const int* __restrict__ meta,
                          int head, bf16* __restrict__ out, int B, int L, int H){
  int idx = blockIdx.x * blockDim.x + threadIdx.x;
  if (idx >= B * L * H) return;
  int h = idx % H;
  int l = (idx / H) % L;
  int b = idx / (H * L);
  const int* m = meta + (head * B + b) * 4;
  bool on = m[2] && (l < m[1]);
  int g = m[0] + l; if (g > L - 1) g = L - 1;
  out[idx] = on ? (bf16)tok[((size_t)b * L + g) * H + h] : (bf16)0.0f;
}

__global__ void hik_combine(const float* __restrict__ P, const float* __restrict__ X,
                            const int* __restrict__ meta, int head,
                            float* __restrict__ out, int B, int L, int H){
  int idx = blockIdx.x * blockDim.x + threadIdx.x;
  if (idx >= B * L * H) return;
  int l = (idx / H) % L;
  int b = idx / (H * L);
  const int* m = meta + (head * B + b) * 4;
  float mask = (m[2] && l < m[1]) ? 1.f : 0.f;
  out[idx] = P[idx] * mask + X[idx];
}

// v[b,l,r] = sum_h tanh(px[b,l,h]+pg[b,h]+pr[r,h]) * vw[h] + vb
__global__ void attn_scores(const float* __restrict__ px, const float* __restrict__ pg,
                            const float* __restrict__ pr, const float* __restrict__ vw,
                            const float* __restrict__ vb, float* __restrict__ vout,
                            int L, int R, int H){
  __shared__ float s_x[768];
  __shared__ float s_vw[768];
  int row = blockIdx.x;           // b*L + l
  int b   = row / L;
  for (int h = threadIdx.x; h < H; h += blockDim.x){
    s_x[h]  = px[(size_t)row * H + h] + pg[(size_t)b * H + h];
    s_vw[h] = vw[h];
  }
  __syncthreads();
  int lane = threadIdx.x & 31, wave = threadIdx.x >> 5;
  for (int r = wave; r < R; r += 8){
    float s = 0.f;
    for (int h = lane; h < H; h += 32)
      s += tanhf(s_x[h] + pr[(size_t)r * H + h]) * s_vw[h];
    s = wsum(s);
    if (lane == 0) vout[(size_t)row * R + r] = s + vb[0];
  }
}

// softmax over L per (b,r); writes bf16 transposed At[b][r*L + l]  (L == 128)
__global__ void softmax_L(const float* __restrict__ v, bf16* __restrict__ At,
                          int B, int L, int R){
  int wv = (int)((blockIdx.x * blockDim.x + threadIdx.x) >> 5);
  int lane = threadIdx.x & 31;
  if (wv >= B * R) return;
  int b = wv / R, r = wv % R;
  float vals[4], m = -1e30f;
#pragma unroll
  for (int i = 0; i < 4; ++i){
    int l = lane + 32 * i;
    vals[i] = v[((size_t)b * L + l) * R + r];
    m = fmaxf(m, vals[i]);
  }
  m = wmax(m);
  float s = 0.f;
#pragma unroll
  for (int i = 0; i < 4; ++i){ vals[i] = __expf(vals[i] - m); s += vals[i]; }
  s = wsum(s);
  float inv = 1.f / s;
#pragma unroll
  for (int i = 0; i < 4; ++i)
    At[((size_t)b * R + r) * L + lane + 32 * i] = (bf16)(vals[i] * inv);
}

__global__ void bcast_add(const float* __restrict__ u, const float* __restrict__ c,
                          const float* __restrict__ bias, float* __restrict__ out,
                          int B, int L, int R){
  int idx = blockIdx.x * blockDim.x + threadIdx.x;
  if (idx >= B * L * R) return;
  int r = idx % R;
  int l = (idx / R) % L;
  int b = idx / (L * R);
  out[idx] = u[b * L + l] + c[b * R + r] + bias[0];
}

// ---------------- host ----------------
extern "C" void kernel_launch(void* const* d_in, const int* in_sizes, int n_in,
                              void* d_out, int out_size, void* d_ws, size_t ws_size,
                              hipStream_t stream) {
  constexpr int B = 4, L = 128, R = 48, H = 768, D = 100, BL = B * L;
  constexpr int BP = 16;  // h_gs rows padded to one full 16-row tile

  const float* h_gs    = (const float*)d_in[0];
  const float* tok     = (const float*)d_in[1];
  const float* f_rel   = (const float*)d_in[2];
  const float* b_rel_t = (const float*)d_in[3];
  const float* rproj_w = (const float*)d_in[4];
  const float* rproj_b = (const float*)d_in[5];
  const float* fcw     = (const float*)d_in[6];
  const float* fcb     = (const float*)d_in[7];
  const float* bigw    = (const float*)d_in[8];
  const float* bigb    = (const float*)d_in[9];
  float* out = (float*)d_out;

  // output offsets (floats), reference return order
  float* o_fhs = out + 0;
  float* o_fhe = out + 512;
  float* o_fts = out + 1024;
  float* o_fte = out + 25600;
  float* o_bts = out + 50176;
  float* o_bte = out + 50688;
  float* o_bhs = out + 51200;
  float* o_bhe = out + 75776;

  // workspace carve-up
  char* ws = (char*)d_ws;
  size_t off = 0;
  auto alloc = [&](size_t bytes) -> char* {
    char* p = ws + off;
    off += (bytes + 255) & ~(size_t)255;
    return p;
  };
  bf16*  wt     = (bf16*)alloc((size_t)10 * H * H * 2);
  bf16*  tokb   = (bf16*)alloc((size_t)BL * H * 2);
  bf16*  fpadb  = (bf16*)alloc((size_t)BL * H * 2);
  bf16*  bpadb  = (bf16*)alloc((size_t)BL * H * 2);
  bf16*  frelb  = (bf16*)alloc((size_t)R * H * 2);
  bf16*  brelb  = (bf16*)alloc((size_t)R * H * 2);
  bf16*  hgsb   = (bf16*)alloc((size_t)BP * H * 2);   // zero-padded to 16 rows
  bf16*  tokT   = (bf16*)alloc((size_t)B * H * L * 2);
  float* brel32 = (float*)alloc((size_t)R * H * 4);
  float* Pfhw   = (float*)alloc((size_t)BL * H * 4);
  float* Pbtw   = (float*)alloc((size_t)BL * H * 4);
  float* prf    = (float*)alloc((size_t)R * H * 4);
  float* prb    = (float*)alloc((size_t)R * H * 4);
  float* pgf    = (float*)alloc((size_t)BP * H * 4);  // padded output rows
  float* pgb    = (float*)alloc((size_t)BP * H * 4);
  float* pxf    = (float*)alloc((size_t)BL * H * 4);
  float* pxb    = (float*)alloc((size_t)BL * H * 4);
  float* X8     = (float*)alloc((size_t)BL * H * 4);
  float* X9     = (float*)alloc((size_t)BL * H * 4);
  float* Hikf   = (float*)alloc((size_t)BL * H * 4);
  float* Hikb   = (float*)alloc((size_t)BL * H * 4);
  float* vf     = (float*)alloc((size_t)BL * R * 4);
  float* vb_    = (float*)alloc((size_t)BL * R * 4);
  bf16*  Atf    = (bf16*)alloc((size_t)B * R * L * 2);
  bf16*  Atb    = (bf16*)alloc((size_t)B * R * L * 2);
  float* Cf     = (float*)alloc((size_t)B * R * H * 4);
  float* Cb     = (float*)alloc((size_t)B * R * H * 4);
  float* u_fs   = (float*)alloc(BL * 4);
  float* u_fe   = (float*)alloc(BL * 4);
  float* u_bs   = (float*)alloc(BL * 4);
  float* u_be   = (float*)alloc(BL * 4);
  float* c_fs   = (float*)alloc(B * R * 4);
  float* c_fe   = (float*)alloc(B * R * 4);
  float* c_bs   = (float*)alloc(B * R * 4);
  float* c_be   = (float*)alloc(B * R * 4);
  int*   meta   = (int*)alloc(8 * 4 * 4);

  const int TB = 256;
  auto blocks = [](int n, int t){ return (n + t - 1) / t; };

  // 1) fc boundary heads: wave per row
  {
    dim3 g(blocks(BL * 32, TB));
    rowdot<<<g, TB, 0, stream>>>(tok, fcw + 0 * H, fcb + 0, o_fhs, BL, H);
    rowdot<<<g, TB, 0, stream>>>(tok, fcw + 1 * H, fcb + 1, o_fhe, BL, H);
    rowdot<<<g, TB, 0, stream>>>(tok, fcw + 4 * H, fcb + 4, o_bts, BL, H);
    rowdot<<<g, TB, 0, stream>>>(tok, fcw + 5 * H, fcb + 5, o_bte, BL, H);
  }
  // 2) b_rel_embs (f32, K=100)
  rel_proj<<<blocks(R * H, TB), TB, 0, stream>>>(b_rel_t, rproj_w, rproj_b, brel32, R, D, H);

  // 3) bf16 conversions / transposes
  to_bf16<<<blocks(BL * H, TB), TB, 0, stream>>>(tok, tokb, BL * H);
  to_bf16<<<blocks(R * H, TB), TB, 0, stream>>>(f_rel, frelb, R * H);
  to_bf16_pad<<<blocks(BP * H, TB), TB, 0, stream>>>(h_gs, hgsb, B * H, BP * H);
  to_bf16<<<blocks(R * H, TB), TB, 0, stream>>>(brel32, brelb, R * H);
  { dim3 g(blocks(H * H, TB), 10); wt_bf16<<<g, TB, 0, stream>>>(bigw, wt, H); }
  tokT_bf16_k<<<blocks(B * H * L, TB), TB, 0, stream>>>(tok, tokT, B, L, H);

  // 4) span extraction (reads the fc-head logits just written)
  span_extract<<<1, 8, 0, stream>>>(o_fhs, o_fhe, o_bts, o_bte, meta, B, L);

  // 5) padded span matrices (bf16, masked gather)
  pad_build<<<blocks(BL * H, TB), TB, 0, stream>>>(tok, meta, 0, fpadb, B, L, H);
  pad_build<<<blocks(BL * H, TB), TB, 0, stream>>>(tok, meta, 1, bpadb, B, L, H);

  // 6) WMMA GEMMs:  Out = A * big_w[j] + big_b[j]   (16x64 tile per wave)
  auto gemm = [&](const bf16* A, int j, float* O, int M){
    int tiles = (M / 16) * (H / 64);
    dim3 g(blocks(tiles, 8), 1);
    wmma_gemm4<<<g, TB, 0, stream>>>(A, wt + (size_t)j * H * H, bigb + (size_t)j * H,
                                     O, M, H, H, 0, 0, 0);
  };
  gemm(fpadb, 0, Pfhw, BL);
  gemm(bpadb, 1, Pbtw, BL);
  gemm(frelb, 2, prf,  R);
  gemm(hgsb,  3, pgf,  BP);
  gemm(tokb,  4, pxf,  BL);
  gemm(brelb, 5, prb,  R);
  gemm(hgsb,  6, pgb,  BP);
  gemm(tokb,  7, pxb,  BL);
  gemm(tokb,  8, X8,   BL);
  gemm(tokb,  9, X9,   BL);

  // 7) Hik = P*mask + X
  hik_combine<<<blocks(BL * H, TB), TB, 0, stream>>>(Pfhw, X8, meta, 0, Hikf, B, L, H);
  hik_combine<<<blocks(BL * H, TB), TB, 0, stream>>>(Pbtw, X9, meta, 1, Hikb, B, L, H);

  // 8) attention scores (tanh energy), per (b,l) block
  attn_scores<<<BL, TB, 0, stream>>>(pxf, pgf, prf, fcw + 8 * H, fcb + 8, vf, L, R, H);
  attn_scores<<<BL, TB, 0, stream>>>(pxb, pgb, prb, fcw + 8 * H, fcb + 8, vb_, L, R, H);

  // 9) softmax over L -> bf16 A^T
  {
    dim3 g(blocks(B * R * 32, TB));
    softmax_L<<<g, TB, 0, stream>>>(vf, Atf, B, L, R);
    softmax_L<<<g, TB, 0, stream>>>(vb_, Atb, B, L, R);
  }

  // 10) context C[b] = At[b](48x128) * tokT[b](768x128)^T   (batched WMMA)
  {
    int tiles = (R / 16) * (H / 64);
    dim3 g(blocks(tiles, 8), B);
    wmma_gemm4<<<g, TB, 0, stream>>>(Atf, tokT, nullptr, Cf, R, H, L,
                                     (long)R * L, (long)H * L, (long)R * H);
    wmma_gemm4<<<g, TB, 0, stream>>>(Atb, tokT, nullptr, Cb, R, H, L,
                                     (long)R * L, (long)H * L, (long)R * H);
  }

  // 11) span-logit dot products
  {
    dim3 gu(blocks(BL * 32, TB));
    rowdot2<<<gu, TB, 0, stream>>>(Hikf, tok, fcw + 2 * H, u_fs, BL, H);
    rowdot2<<<gu, TB, 0, stream>>>(Hikf, tok, fcw + 3 * H, u_fe, BL, H);
    rowdot2<<<gu, TB, 0, stream>>>(Hikb, tok, fcw + 6 * H, u_bs, BL, H);
    rowdot2<<<gu, TB, 0, stream>>>(Hikb, tok, fcw + 7 * H, u_be, BL, H);
    dim3 gc(blocks(B * R * 32, TB));
    rowdot<<<gc, TB, 0, stream>>>(Cf, fcw + 2 * H, nullptr, c_fs, B * R, H);
    rowdot<<<gc, TB, 0, stream>>>(Cf, fcw + 3 * H, nullptr, c_fe, B * R, H);
    rowdot<<<gc, TB, 0, stream>>>(Cb, fcw + 6 * H, nullptr, c_bs, B * R, H);
    rowdot<<<gc, TB, 0, stream>>>(Cb, fcw + 7 * H, nullptr, c_be, B * R, H);
  }

  // 12) broadcast assembly of (B,L,R) logits
  {
    dim3 g(blocks(B * L * R, TB));
    bcast_add<<<g, TB, 0, stream>>>(u_fs, c_fs, fcb + 2, o_fts, B, L, R);
    bcast_add<<<g, TB, 0, stream>>>(u_fe, c_fe, fcb + 3, o_fte, B, L, R);
    bcast_add<<<g, TB, 0, stream>>>(u_bs, c_bs, fcb + 6, o_bhs, B, L, R);
    bcast_add<<<g, TB, 0, stream>>>(u_be, c_be, fcb + 7, o_bhe, B, L, R);
  }
  (void)in_sizes; (void)n_in; (void)out_size; (void)ws_size;
}
